// transformer_block_26302379721305
// MI455X (gfx1250) — matrix operand order repaired and checked by reading, not compiled, not run on hardware
//
#include <hip/hip_runtime.h>
#include <hip/hip_bf16.h>
#include <math.h>

// ---------------------------------------------------------------------------
// Types for CDNA5 WMMA (wave32): A/B = 16 x bf16 per lane, C/D = 8 x f32.
// ---------------------------------------------------------------------------
typedef __attribute__((ext_vector_type(16))) __bf16 v16bf;
typedef __attribute__((ext_vector_type(8)))  float  v8f;

union FragBF { v16bf v; uint4 q[2]; };
union U8BF   { uint4 q; __bf16 b[8]; };

static __device__ __forceinline__ v8f vzero8() {
  v8f z = {0.f,0.f,0.f,0.f,0.f,0.f,0.f,0.f};
  return z;
}

// ---------------------------------------------------------------------------
// CDNA5 async memory->LDS path (ASYNCcnt-tracked, bypasses the VGPR file).
// LDS aperture lives entirely in addr[63:32], so the low 32 bits of a generic
// pointer to a __shared__ object are exactly the LDS byte offset the
// instruction's VDST operand wants.
// ---------------------------------------------------------------------------
static __device__ __forceinline__ void async_ld16(unsigned lds_off, const void* g) {
  asm volatile("global_load_async_to_lds_b128 %0, %1, off"
               :: "v"(lds_off), "v"(g) : "memory");
}
static __device__ __forceinline__ void wait_async0() {
  asm volatile("s_wait_asynccnt 0" ::: "memory");
}

// ---------------------------------------------------------------------------
// Generic bf16 GEMM: C[M,N] = A[M,K] * B[N,K]^T  (both K-contiguous, row major)
// Block tile 128x128, K-step 32, double-buffered LDS fed by async loads.
// 8 waves; each wave owns a 32x64 tile -> 8 v_wmma_f32_16x16x32_bf16 / K-step.
// EPI: 0 = f32 out, 1 = bf16 out, 2 = f32 out + f32 residual, 3 = bf16 silu
// ---------------------------------------------------------------------------
#define BM 128
#define BN 128
#define BK 32
#define LDSTR 40   // bf16 LDS row stride (32 + 8 pad); 80B rows keep 16B align

template <int EPI>
__global__ __launch_bounds__(256, 2)
void gemm_bf16_wmma(const __bf16* __restrict__ A, const __bf16* __restrict__ B,
                    int M, int N, int Kd,
                    float* __restrict__ Cf, __bf16* __restrict__ Cb,
                    const float* __restrict__ R) {
  __shared__ __bf16 As[2][BM * LDSTR];
  __shared__ __bf16 Bs[2][BN * LDSTR];

  const int tid  = threadIdx.x;
  const int lane = tid & 31;
  const int wid  = tid >> 5;
  const int wm   = wid & 3;          // 4 waves along M (32 rows each)
  const int wn   = wid >> 2;         // 2 waves along N (64 cols each)
  const int hi   = lane >> 4;        // half-wave selector (ISA frag layout)
  const int l15  = lane & 15;
  const int mblk = blockIdx.y * BM;
  const int nblk = blockIdx.x * BN;

  v8f acc[2][4];
#pragma unroll
  for (int mi = 0; mi < 2; mi++)
#pragma unroll
    for (int ni = 0; ni < 4; ni++) acc[mi][ni] = vzero8();

  // stage one 128x32 A tile + 128x32 B tile straight into LDS (async engine)
  auto stage = [&](int buf, int k0) {
#pragma unroll
    for (int i = 0; i < 2; i++) {
      int idx = tid + i * 256;       // 512 uint4 per tile
      int row = idx >> 2;
      int c4  = idx & 3;
      async_ld16((unsigned)(size_t)&As[buf][row * LDSTR + c4 * 8],
                 A + (size_t)(mblk + row) * Kd + k0 + c4 * 8);
      async_ld16((unsigned)(size_t)&Bs[buf][row * LDSTR + c4 * 8],
                 B + (size_t)(nblk + row) * Kd + k0 + c4 * 8);
    }
  };

  const int nk = Kd / BK;
  stage(0, 0);
  wait_async0();
  __syncthreads();

  for (int kt = 0; kt < nk; kt++) {
    const int buf = kt & 1;
    if (kt + 1 < nk) stage(buf ^ 1, (kt + 1) * BK);  // overlap with compute

    // ---- assemble WMMA fragments with contiguous k = hi*16 + s in A and B ----
    FragBF fa[2], fb[4];
#pragma unroll
    for (int mi = 0; mi < 2; mi++) {
      const __bf16* pa = &As[buf][(wm * 32 + mi * 16 + l15) * LDSTR];
      fa[mi].q[0] = *(const uint4*)(pa + hi * 16);       // K = hi*16 .. +7
      fa[mi].q[1] = *(const uint4*)(pa + hi * 16 + 8);   // K = hi*16+8 .. +15
    }
#pragma unroll
    for (int ni = 0; ni < 4; ni++) {
      const __bf16* pb = &Bs[buf][(wn * 64 + ni * 16 + l15) * LDSTR + hi * 16];
      fb[ni].q[0] = *(const uint4*)(pb);                 // K = hi*16 .. +15
      fb[ni].q[1] = *(const uint4*)(pb + 8);
    }

#pragma unroll
    for (int mi = 0; mi < 2; mi++)
#pragma unroll
      for (int ni = 0; ni < 4; ni++)
        acc[mi][ni] = __builtin_amdgcn_wmma_f32_16x16x32_bf16(
            false, fa[mi].v, false, fb[ni].v, (short)0, acc[mi][ni], false, false);

    wait_async0();      // next tile fully resident in the other buffer
    __syncthreads();
  }

  // ---- epilogue: C/D layout row = r + 8*hi, col = l15 ----
#pragma unroll
  for (int mi = 0; mi < 2; mi++)
#pragma unroll
    for (int ni = 0; ni < 4; ni++)
#pragma unroll
      for (int r = 0; r < 8; r++) {
        int row = mblk + wm * 32 + mi * 16 + r + 8 * hi;
        int col = nblk + wn * 64 + ni * 16 + l15;
        size_t off = (size_t)row * N + col;
        float v = acc[mi][ni][r];
        if (EPI == 0) {
          Cf[off] = v;
        } else if (EPI == 1) {
          Cb[off] = (__bf16)v;
        } else if (EPI == 2) {
          Cf[off] = v + R[off];
        } else {                       // SiLU for the w1 branch of SwiGLU
          float s = v / (1.0f + __expf(-v));
          Cb[off] = (__bf16)s;
        }
      }
}

// ---------------------------------------------------------------------------
// Flash attention, one block per (batch, head, 64-query tile). 4 waves.
// S = Q K^T via WMMA (dh=64 -> 2 K-steps), online softmax in registers with
// half-wave shuffles, O += P V via WMMA (V staged transposed in LDS).
// Q/K tiles staged with async loads; V transposed through VGPRs.
// ---------------------------------------------------------------------------
__global__ __launch_bounds__(128, 2)
void attn_fa(const __bf16* __restrict__ Q, const __bf16* __restrict__ K,
             const __bf16* __restrict__ V, __bf16* __restrict__ O, int SL) {
  __shared__ __bf16 Qs[64 * 64];        // [q][dh]
  __shared__ __bf16 Ks[64 * 64];        // [kv][dh]
  __shared__ __bf16 Vs[64 * 64];        // transposed: [dh][kv]
  __shared__ __bf16 Ps[4][16 * 64];     // per-wave P (A-fragment layout source)

  const int qt   = blockIdx.x;
  const int h    = blockIdx.y;
  const int bb   = blockIdx.z;
  const int tid  = threadIdx.x;
  const int lane = tid & 31;
  const int wid  = tid >> 5;
  const int hi   = lane >> 4;
  const int l15  = lane & 15;
  const size_t base = ((size_t)bb * SL) * 1024 + (size_t)h * 64;

  // async-stage Q tile (64x64 bf16); completion covered by in-loop wait
#pragma unroll
  for (int i = 0; i < 4; i++) {
    int idx = tid + i * 128;
    int row = idx >> 3, c8 = (idx & 7) * 8;
    async_ld16((unsigned)(size_t)&Qs[row * 64 + c8],
               Q + base + (size_t)(qt * 64 + row) * 1024 + c8);
  }

  v8f o[4];
  float m[8], l[8];
#pragma unroll
  for (int t = 0; t < 4; t++) o[t] = vzero8();
#pragma unroll
  for (int r = 0; r < 8; r++) { m[r] = -1e30f; l[r] = 0.f; }

  for (int j = 0; j <= qt; j++) {
    __syncthreads();  // previous-iteration readers done before overwriting

    // async-stage K tile [kv][dh]
#pragma unroll
    for (int i = 0; i < 4; i++) {
      int idx = tid + i * 128;
      int row = idx >> 3, c8 = (idx & 7) * 8;
      async_ld16((unsigned)(size_t)&Ks[row * 64 + c8],
                 K + base + (size_t)(j * 64 + row) * 1024 + c8);
    }
    // load V tile transposed into Vs[dh][kv] (needs lane transpose -> VGPRs)
#pragma unroll
    for (int i = 0; i < 4; i++) {
      int idx = tid + i * 128;
      int kv = idx >> 3, c8 = (idx & 7) * 8;
      U8BF u;
      u.q = *(const uint4*)(V + base + (size_t)(j * 64 + kv) * 1024 + c8);
#pragma unroll
      for (int e = 0; e < 8; e++) Vs[(c8 + e) * 64 + kv] = u.b[e];
    }
    wait_async0();    // Q (first iter) + K tiles resident
    __syncthreads();

    // ---- S = Q K^T for this wave's 16 query rows (4 tiles of 16 kv) ----
    v8f s[4];
#pragma unroll
    for (int ni = 0; ni < 4; ni++) s[ni] = vzero8();
#pragma unroll
    for (int ks = 0; ks < 2; ks++) {
      FragBF fq;
      const __bf16* pa = &Qs[(wid * 16 + l15) * 64 + ks * 32];
      fq.q[0] = *(const uint4*)(pa + hi * 16);
      fq.q[1] = *(const uint4*)(pa + hi * 16 + 8);
#pragma unroll
      for (int ni = 0; ni < 4; ni++) {
        FragBF fk;
        const __bf16* pb = &Ks[(ni * 16 + l15) * 64 + ks * 32 + hi * 16];
        fk.q[0] = *(const uint4*)(pb);
        fk.q[1] = *(const uint4*)(pb + 8);
        s[ni] = __builtin_amdgcn_wmma_f32_16x16x32_bf16(
            false, fq.v, false, fk.v, (short)0, s[ni], false, false);
      }
    }

    // ---- scale + causal mask (diagonal tile only) ----
    const bool diag = (j == qt);
#pragma unroll
    for (int ni = 0; ni < 4; ni++)
#pragma unroll
      for (int r = 0; r < 8; r++) {
        float t = s[ni][r] * 0.125f;   // 1/sqrt(64)
        if (diag && (ni * 16 + l15) > (r + 8 * hi)) t = -1e30f;
        s[ni][r] = t;
      }

    // ---- online softmax; rows live in half-waves (shfl_xor masks 1..8) ----
#pragma unroll
    for (int r = 0; r < 8; r++) {
      float mx = fmaxf(fmaxf(s[0][r], s[1][r]), fmaxf(s[2][r], s[3][r]));
#pragma unroll
      for (int d = 1; d < 16; d <<= 1) mx = fmaxf(mx, __shfl_xor(mx, d, 32));
      float mc  = fmaxf(m[r], mx);
      float scl = __expf(m[r] - mc);
      m[r] = mc;
      float rs = 0.f;
#pragma unroll
      for (int ni = 0; ni < 4; ni++) {
        float pv = __expf(s[ni][r] - mc);
        s[ni][r] = pv;
        rs += pv;
      }
#pragma unroll
      for (int d = 1; d < 16; d <<= 1) rs += __shfl_xor(rs, d, 32);
      l[r] = l[r] * scl + rs;
      // rescale ONLY row r (vector element r) of each dh-tile accumulator
#pragma unroll
      for (int t = 0; t < 4; t++) o[t][r] *= scl;
      // write P row to per-wave LDS in A-matrix source order [m][k]
#pragma unroll
      for (int ni = 0; ni < 4; ni++)
        Ps[wid][(r + 8 * hi) * 64 + ni * 16 + l15] = (__bf16)s[ni][r];
    }

    // ---- O += P V  (K dim = 64 kv -> 2 WMMA steps, 4 dh tiles) ----
#pragma unroll
    for (int ks = 0; ks < 2; ks++) {
      FragBF fp;
      const __bf16* pa = &Ps[wid][l15 * 64 + ks * 32];
      fp.q[0] = *(const uint4*)(pa + hi * 16);
      fp.q[1] = *(const uint4*)(pa + hi * 16 + 8);
#pragma unroll
      for (int ni = 0; ni < 4; ni++) {
        FragBF fv;
        const __bf16* pb = &Vs[(ni * 16 + l15) * 64 + ks * 32 + hi * 16];
        fv.q[0] = *(const uint4*)(pb);
        fv.q[1] = *(const uint4*)(pb + 8);
        o[ni] = __builtin_amdgcn_wmma_f32_16x16x32_bf16(
            false, fp.v, false, fv.v, (short)0, o[ni], false, false);
      }
    }
  }

  // ---- normalize and write context (bf16, [token, h*64+dh]) ----
#pragma unroll
  for (int ni = 0; ni < 4; ni++)
#pragma unroll
    for (int r = 0; r < 8; r++) {
      int qrow = qt * 64 + wid * 16 + r + 8 * hi;
      size_t off = base + (size_t)qrow * 1024 + ni * 16 + l15;
      O[off] = (__bf16)(o[ni][r] / l[r]);
    }
}

// ---------------------------------------------------------------------------
// RMSNorm (d=1024): one block per token, outputs bf16 (xn * w).
// ---------------------------------------------------------------------------
__global__ __launch_bounds__(256)
void rmsnorm_bf16_k(const float* __restrict__ X, const float* __restrict__ W,
                    __bf16* __restrict__ Y) {
  __shared__ float red[8];
  const int t = blockIdx.x;
  const float4 v = ((const float4*)(X + (size_t)t * 1024))[threadIdx.x];
  float ss = v.x * v.x + v.y * v.y + v.z * v.z + v.w * v.w;
#pragma unroll
  for (int d = 1; d < 32; d <<= 1) ss += __shfl_xor(ss, d, 32);
  if ((threadIdx.x & 31) == 0) red[threadIdx.x >> 5] = ss;
  __syncthreads();
  float tot = 0.f;
#pragma unroll
  for (int i = 0; i < 8; i++) tot += red[i];
  const float rinv = rsqrtf(tot * (1.0f / 1024.0f) + 1e-5f);
  const float4 w4 = ((const float4*)W)[threadIdx.x];
  __bf16* y = Y + (size_t)t * 1024 + threadIdx.x * 4;
  y[0] = (__bf16)(v.x * rinv * w4.x);
  y[1] = (__bf16)(v.y * rinv * w4.y);
  y[2] = (__bf16)(v.z * rinv * w4.z);
  y[3] = (__bf16)(v.w * rinv * w4.w);
}

// ---------------------------------------------------------------------------
// RoPE on Q and K in place (bf16), interleaved (even,odd) pairs, dh=64.
// ---------------------------------------------------------------------------
__global__ void rope_k(__bf16* __restrict__ Q, __bf16* __restrict__ Kq,
                       int SL, int T) {
  int gid = blockIdx.x * blockDim.x + threadIdx.x;  // T * 16 heads * 32 pairs
  if (gid >= T * 512) return;
  int t = gid >> 9, r = gid & 511;
  int h = r >> 5, j = r & 31;
  int pos = t % SL;
  // inv_freq = 10000^(-2j/64) = exp(-j * ln(10000)/32)
  float ang = (float)pos * __expf(-(float)j * 0.28782313662425f);
  float sn, cs;
  __sincosf(ang, &sn, &cs);
  size_t off = (size_t)t * 1024 + h * 64 + 2 * j;
  float qe = (float)Q[off], qo = (float)Q[off + 1];
  Q[off]     = (__bf16)(cs * qe - sn * qo);
  Q[off + 1] = (__bf16)(sn * qe + cs * qo);
  float ke = (float)Kq[off], ko = (float)Kq[off + 1];
  Kq[off]     = (__bf16)(cs * ke - sn * ko);
  Kq[off + 1] = (__bf16)(sn * ke + cs * ko);
}

// ---------------------------------------------------------------------------
// f32 -> bf16 convert (weights), 4 elements per thread.
// ---------------------------------------------------------------------------
__global__ void f2bf_k(const float* __restrict__ X, __bf16* __restrict__ Y,
                       size_t n4) {
  size_t i = (size_t)blockIdx.x * blockDim.x + threadIdx.x;
  if (i >= n4) return;
  float4 v = ((const float4*)X)[i];
  __bf16* y = Y + 4 * i;
  y[0] = (__bf16)v.x; y[1] = (__bf16)v.y; y[2] = (__bf16)v.z; y[3] = (__bf16)v.w;
}

// ---------------------------------------------------------------------------
// SwiGLU gate multiply: A (already SiLU'd) *= B, bf16, in place.
// ---------------------------------------------------------------------------
__global__ void mulbf_k(__bf16* __restrict__ A, const __bf16* __restrict__ B,
                        size_t n4) {
  size_t i = (size_t)blockIdx.x * blockDim.x + threadIdx.x;
  if (i >= n4) return;
  __bf16* a = A + 4 * i;
  const __bf16* b = B + 4 * i;
#pragma unroll
  for (int e = 0; e < 4; e++) a[e] = (__bf16)((float)a[e] * (float)b[e]);
}

// ---------------------------------------------------------------------------
// Host-side launch
// ---------------------------------------------------------------------------
extern "C" void kernel_launch(void* const* d_in, const int* in_sizes, int n_in,
                              void* d_out, int out_size, void* d_ws, size_t ws_size,
                              hipStream_t stream) {
  const float* x   = (const float*)d_in[0];
  const float* wq  = (const float*)d_in[1];
  const float* wk  = (const float*)d_in[2];
  const float* wv  = (const float*)d_in[3];
  const float* wo  = (const float*)d_in[4];
  const float* ln1 = (const float*)d_in[5];
  const float* ln2 = (const float*)d_in[6];
  const float* w1  = (const float*)d_in[7];
  const float* w2  = (const float*)d_in[8];
  const float* w3  = (const float*)d_in[9];

  const int D = 1024, DFF = 4096, H = 16;
  const int T  = in_sizes[0] / D;          // b * sl
  const int SL = (T >= 2048) ? 2048 : T;   // reference shape: sl = 2048
  const int NB = T / SL;

  char* p = (char*)d_ws;
  auto carve = [&](size_t bytes) -> char* {
    char* r = p;
    p += (bytes + 255) & ~(size_t)255;
    return r;
  };
  __bf16* wqb  = (__bf16*)carve((size_t)D * D * 2);
  __bf16* wkb  = (__bf16*)carve((size_t)D * D * 2);
  __bf16* wvb  = (__bf16*)carve((size_t)D * D * 2);
  __bf16* wob  = (__bf16*)carve((size_t)D * D * 2);
  __bf16* w1b  = (__bf16*)carve((size_t)DFF * D * 2);
  __bf16* w3b  = (__bf16*)carve((size_t)DFF * D * 2);
  __bf16* w2b  = (__bf16*)carve((size_t)D * DFF * 2);
  __bf16* xnb  = (__bf16*)carve((size_t)T * D * 2);
  __bf16* qb   = (__bf16*)carve((size_t)T * D * 2);
  __bf16* kb   = (__bf16*)carve((size_t)T * D * 2);
  __bf16* vb   = (__bf16*)carve((size_t)T * D * 2);
  __bf16* ctxb = (__bf16*)carve((size_t)T * D * 2);
  float*  x1f  = (float*) carve((size_t)T * D * 4);
  __bf16* affb = (__bf16*)carve((size_t)T * DFF * 2);
  __bf16* bffb = (__bf16*)carve((size_t)T * DFF * 2);

  const int thr = 256;
  const size_t nW4 = (size_t)D * D / 4;
  const size_t nF4 = (size_t)DFF * D / 4;
  f2bf_k<<<(unsigned)((nW4 + thr - 1) / thr), thr, 0, stream>>>(wq, wqb, nW4);
  f2bf_k<<<(unsigned)((nW4 + thr - 1) / thr), thr, 0, stream>>>(wk, wkb, nW4);
  f2bf_k<<<(unsigned)((nW4 + thr - 1) / thr), thr, 0, stream>>>(wv, wvb, nW4);
  f2bf_k<<<(unsigned)((nW4 + thr - 1) / thr), thr, 0, stream>>>(wo, wob, nW4);
  f2bf_k<<<(unsigned)((nF4 + thr - 1) / thr), thr, 0, stream>>>(w1, w1b, nF4);
  f2bf_k<<<(unsigned)((nF4 + thr - 1) / thr), thr, 0, stream>>>(w3, w3b, nF4);
  f2bf_k<<<(unsigned)((nF4 + thr - 1) / thr), thr, 0, stream>>>(w2, w2b, nF4);

  // --- attention sublayer ---
  rmsnorm_bf16_k<<<T, 256, 0, stream>>>(x, ln1, xnb);
  gemm_bf16_wmma<1><<<dim3(D / 128, T / 128), 256, 0, stream>>>(
      xnb, wqb, T, D, D, nullptr, qb, nullptr);
  gemm_bf16_wmma<1><<<dim3(D / 128, T / 128), 256, 0, stream>>>(
      xnb, wkb, T, D, D, nullptr, kb, nullptr);
  gemm_bf16_wmma<1><<<dim3(D / 128, T / 128), 256, 0, stream>>>(
      xnb, wvb, T, D, D, nullptr, vb, nullptr);
  rope_k<<<(T * 512 + thr - 1) / thr, thr, 0, stream>>>(qb, kb, SL, T);
  attn_fa<<<dim3(SL / 64, H, NB), 128, 0, stream>>>(qb, kb, vb, ctxb, SL);
  gemm_bf16_wmma<2><<<dim3(D / 128, T / 128), 256, 0, stream>>>(
      ctxb, wob, T, D, D, x1f, nullptr, x);

  // --- SwiGLU FFN sublayer ---
  rmsnorm_bf16_k<<<T, 256, 0, stream>>>(x1f, ln2, xnb);
  gemm_bf16_wmma<3><<<dim3(DFF / 128, T / 128), 256, 0, stream>>>(
      xnb, w1b, T, DFF, D, nullptr, affb, nullptr);   // SiLU fused
  gemm_bf16_wmma<1><<<dim3(DFF / 128, T / 128), 256, 0, stream>>>(
      xnb, w3b, T, DFF, D, nullptr, bffb, nullptr);
  const size_t nG4 = (size_t)T * DFF / 4;
  mulbf_k<<<(unsigned)((nG4 + thr - 1) / thr), thr, 0, stream>>>(affb, bffb, nG4);
  gemm_bf16_wmma<2><<<dim3(D / 128, T / 128), 256, 0, stream>>>(
      affb, w2b, T, D, DFF, (float*)d_out, nullptr, x1f);
}
